// MultipleModels_7447473292058
// MI455X (gfx1250) — compile-verified, matrix operand order checked by reading, not run on hardware
//
#include <hip/hip_runtime.h>
#include <hip/hip_bf16.h>

// ---------------------------------------------------------------------------
// MI455X (gfx1250) routed NeRF-ensemble MLP, register-resident activations.
// D = A(W^T tile) x B(activations, N=points). The f32 D layout of two adjacent
// 16-dim tiles equals the bf16 B-fragment layout of the next layer lane-for-
// lane, so activations never touch LDS: bias+relu+cvt_pk_bf16 chains layers
// entirely in VGPRs. Compute-bound (~58 GMACs after routing) on the WMMA pipe.
// ---------------------------------------------------------------------------

typedef __bf16 bf16_t;
typedef __attribute__((ext_vector_type(8)))  __bf16 v8bf;
typedef __attribute__((ext_vector_type(16))) __bf16 v16bf;
typedef __attribute__((ext_vector_type(8)))  float  v8f;

#define N_PTS   131072
#define RES_MAP 1024
#define TILE    128
#define PACK_PER_MODEL 600064   // packed bf16 elems per model

// packed per-model layer offsets (elements)
#define OFF_L0    0
#define OFF_H(j)  (16384 + (j)*65536)   // j = 0..3 -> layers 1..4
#define OFF_SKIP  278528
#define OFF_H4    360448
#define OFF_H5    425984
#define OFF_FEAT  491520
#define OFF_ALPHA 557056
#define OFF_VIEWS 561152
#define OFF_RGB   598016

// ---------------------------------------------------------------------------
__global__ void init_counts(int* counts) {
    if (threadIdx.x < 8) counts[threadIdx.x] = 0;
}

__global__ void route_zero(const float* __restrict__ pts,
                           const int* __restrict__ himg,
                           float* __restrict__ out,
                           int* __restrict__ counts,
                           int* __restrict__ lists) {
    int n = blockIdx.x * 256 + threadIdx.x;
    if (n >= N_PTS) return;
    float x = pts[n * 3 + 0];
    float y = pts[n * 3 + 1];
    int xi = (int)((y + 1.0f) * 0.5f * (float)RES_MAP);
    int yi = (int)((x + 1.0f) * 0.5f * (float)RES_MAP);
    int idx = himg[xi * RES_MAP + yi];
    float4 z = make_float4(0.f, 0.f, 0.f, 0.f);
    *(float4*)(out + (size_t)n * 4) = z;       // model-0 / default contribution
    if (idx > 0) {
        int pos = atomicAdd(&counts[idx], 1);
        lists[(size_t)idx * N_PTS + pos] = n;
    }
}

// Pack fp32 [Ksrc, Ncols] weights (per model) into bf16 WMMA fragments.
// The per-lane K pattern of A and B operands is identical and
// W^T[M,K] == W[K, N=M], so the same packing serves the A-operand (W^T) form:
//   frag = e>>9, lane = (e>>4)&31, j = e&15
//   outdim = ct*16 + (lane&15)
//   k      = kt*32 + (lane>>4)*8 + (j&7) + ((j&8)?16:0)
// remap==1: skip-layer concat [emb51 | pad13 | h256] -> w_skip rows
__global__ void pack_kernel(const float* __restrict__ src, bf16_t* __restrict__ dst,
                            long srcModelStride, long dstModelStride, int elemsPerModel,
                            int Ksrc, int Ncols, int KT, int remap) {
    int e = blockIdx.x * 256 + threadIdx.x;
    if (e >= elemsPerModel) return;
    int m    = blockIdx.y;
    int j    = e & 15;
    int lane = (e >> 4) & 31;
    int frag = e >> 9;
    int kt   = frag % KT;
    int ct   = frag / KT;
    int n    = ct * 16 + (lane & 15);
    int k    = kt * 32 + ((lane >> 4) * 8) + (j & 7) + ((j & 8) ? 16 : 0);
    int kr = k;
    if (remap == 1) kr = (k < 51) ? k : ((k < 64) ? -1 : (k - 13));
    float v = 0.f;
    if (kr >= 0 && kr < Ksrc && n < Ncols)
        v = src[(long)m * srcModelStride + (long)kr * Ncols + n];
    dst[(long)m * dstModelStride + e] = (bf16_t)v;
}

// ---------------------------------------------------------------------------
__device__ __forceinline__ v8f wmma_acc(v16bf a, v16bf b, v8f c) {
    return __builtin_amdgcn_wmma_f32_16x16x32_bf16(false, a, false, b, (short)0, c,
                                                   false, false);
}

// Load one B-fragment (K=32, N=16 points) from an LDS row-major tile.
// lane&15 = point row; elems 0..7 <- [kloc+hl*8 .. +7], 8..15 <- +16.
__device__ __forceinline__ v16bf lds_bfrag(const bf16_t* rowBase, int kloc, int hl) {
    const bf16_t* p = rowBase + kloc + hl * 8;
    v8bf lo = *(const v8bf*)(p);
    v8bf hi = *(const v8bf*)(p + 16);
    return __builtin_shufflevector(lo, hi, 0,1,2,3,4,5,6,7,8,9,10,11,12,13,14,15);
}

// Dense layer: K = KT*32 -> NP*32 output dims, activations in registers.
// frag(kt) yields the kt'th input B-fragment (fully unrolled -> registers).
template <int KT, int NP, class F>
__device__ __forceinline__ void dense_layer(F&& frag, const bf16_t* wfrag,
                                            const float* bias, bool relu,
                                            v16bf (&outF)[NP], int lane) {
    const int hl = lane >> 4;
#pragma unroll
    for (int i = 0; i < NP; ++i) {
        if (i + 1 < NP)
            __builtin_prefetch((const void*)((const v16bf*)wfrag +
                               (((2 * i + 2) * KT) * 32 + lane)), 0, 3);
        v8f a0 = {0.f,0.f,0.f,0.f,0.f,0.f,0.f,0.f};
        v8f a1 = {0.f,0.f,0.f,0.f,0.f,0.f,0.f,0.f};
#pragma unroll
        for (int kt = 0; kt < KT; ++kt) {
            v16bf w0 = *((const v16bf*)wfrag + (((2 * i    ) * KT + kt) * 32 + lane));
            v16bf w1 = *((const v16bf*)wfrag + (((2 * i + 1) * KT + kt) * 32 + lane));
            v16bf bf = frag(kt);
            a0 = wmma_acc(w0, bf, a0);      // two independent D-chains for ILP
            a1 = wmma_acc(w1, bf, a1);
        }
        // bias + relu + pack: D[2i],D[2i+1] -> next-layer B fragment (lane-local)
        const float* bp = bias + i * 32 + hl * 8;
        float4 c0 = *(const float4*)(bp);
        float4 c1 = *(const float4*)(bp + 4);
        float4 c2 = *(const float4*)(bp + 16);
        float4 c3 = *(const float4*)(bp + 20);
        float bb0[8] = {c0.x,c0.y,c0.z,c0.w,c1.x,c1.y,c1.z,c1.w};
        float bb1[8] = {c2.x,c2.y,c2.z,c2.w,c3.x,c3.y,c3.z,c3.w};
        v16bf r;
#pragma unroll
        for (int v = 0; v < 8; ++v) {
            float x = a0[v] + bb0[v];
            if (relu) x = fmaxf(x, 0.f);
            r[v] = (bf16_t)x;
        }
#pragma unroll
        for (int v = 0; v < 8; ++v) {
            float x = a1[v] + bb1[v];
            if (relu) x = fmaxf(x, 0.f);
            r[8 + v] = (bf16_t)x;
        }
        outF[i] = r;
    }
}

// Head: single 16-dim output tile (alpha pad-16 / rgb pad-16), raw f32 acc.
template <int KT, class F>
__device__ __forceinline__ v8f head_tile(F&& frag, const bf16_t* wfrag, int lane) {
    v8f a = {0.f,0.f,0.f,0.f,0.f,0.f,0.f,0.f};
#pragma unroll
    for (int kt = 0; kt < KT; ++kt)
        a = wmma_acc(*((const v16bf*)wfrag + (kt * 32 + lane)), frag(kt), a);
    return a;
}

// ---------------------------------------------------------------------------
__global__ __launch_bounds__(256) void mlp_kernel(
    const float* __restrict__ pts, const float* __restrict__ dirs,
    const bf16_t* __restrict__ packed,
    const float* __restrict__ b_in,   const float* __restrict__ b_hid,
    const float* __restrict__ b_skip, const float* __restrict__ b_feat,
    const float* __restrict__ b_alpha,const float* __restrict__ b_views,
    const float* __restrict__ b_rgb,
    const int* __restrict__ counts, const int* __restrict__ lists,
    float* __restrict__ out) {
    const int m = blockIdx.y + 1;                 // model 0 contributes zero: skipped
    const int cnt = counts[m];
    const int tileStart = blockIdx.x * TILE;
    if (tileStart >= cnt) return;

    extern __shared__ char smemRaw[];
    bf16_t* sEmb  = (bf16_t*)smemRaw;             // [128][64]
    bf16_t* sDirs = sEmb + 128 * 64;              // [128][32]

    // ---- gather + positional encoding (threads 0..127: pts, 128..255: dirs)
    {
        const int t = threadIdx.x;
        const int r = t & 127;
        const int part = t >> 7;
        const int g = tileStart + r;
        const int n = (g < cnt) ? lists[(size_t)m * N_PTS + g] : -1;
        const float* src = part ? dirs : pts;
        float v0 = 0.f, v1 = 0.f, v2 = 0.f;
        if (n >= 0) { v0 = src[n*3+0]; v1 = src[n*3+1]; v2 = src[n*3+2]; }
        if (part == 0) {
            bf16_t* e = sEmb + r * 64;
            e[0] = (bf16_t)v0; e[1] = (bf16_t)v1; e[2] = (bf16_t)v2;
#pragma unroll
            for (int f = 0; f < 8; ++f) {
                float fr = (float)(1 << f);
                e[3+f*6+0] = (bf16_t)__sinf(v0*fr); e[3+f*6+1] = (bf16_t)__sinf(v1*fr); e[3+f*6+2] = (bf16_t)__sinf(v2*fr);
                e[3+f*6+3] = (bf16_t)__cosf(v0*fr); e[3+f*6+4] = (bf16_t)__cosf(v1*fr); e[3+f*6+5] = (bf16_t)__cosf(v2*fr);
            }
#pragma unroll
            for (int c = 51; c < 64; ++c) e[c] = (bf16_t)0.f;
        } else {
            bf16_t* e = sDirs + r * 32;
            e[0] = (bf16_t)v0; e[1] = (bf16_t)v1; e[2] = (bf16_t)v2;
#pragma unroll
            for (int f = 0; f < 3; ++f) {
                float fr = (float)(1 << f);
                e[3+f*6+0] = (bf16_t)__sinf(v0*fr); e[3+f*6+1] = (bf16_t)__sinf(v1*fr); e[3+f*6+2] = (bf16_t)__sinf(v2*fr);
                e[3+f*6+3] = (bf16_t)__cosf(v0*fr); e[3+f*6+4] = (bf16_t)__cosf(v1*fr); e[3+f*6+5] = (bf16_t)__cosf(v2*fr);
            }
#pragma unroll
            for (int c = 21; c < 32; ++c) e[c] = (bf16_t)0.f;
        }
    }
    __syncthreads();   // after this, each wave touches only its own 16 rows

    const int lane = threadIdx.x & 31;
    const int hl   = lane >> 4;
    const int rowL = lane & 15;
    const int wrb  = (threadIdx.x >> 5) * 16;     // wave row base (16 points/wave)
    const int myRow = wrb + rowL;
    const int g = tileStart + myRow;
    const int myN = (g < cnt) ? lists[(size_t)m * N_PTS + g] : -1;

    // embedding fragments (only LDS traffic in the whole MLP)
    v16bf embF0 = lds_bfrag(sEmb + myRow * 64, 0, hl);
    v16bf embF1 = lds_bfrag(sEmb + myRow * 64, 32, hl);
    v16bf dirF  = lds_bfrag(sDirs + myRow * 32, 0, hl);

    const bf16_t* W = packed + (size_t)m * PACK_PER_MODEL;
    v16bf hA[8], hB[8], vout[4];

    // L0: emb(K=64) -> 256
    dense_layer<2, 8>([&](int kt) { return kt == 0 ? embF0 : embF1; },
                      W + OFF_L0, b_in + m*256, true, hA, lane);
    // L1..L4
    dense_layer<8, 8>([&](int kt) { return hA[kt]; }, W + OFF_H(0), b_hid + m*1536 + 0*256, true, hB, lane);
    dense_layer<8, 8>([&](int kt) { return hB[kt]; }, W + OFF_H(1), b_hid + m*1536 + 1*256, true, hA, lane);
    dense_layer<8, 8>([&](int kt) { return hA[kt]; }, W + OFF_H(2), b_hid + m*1536 + 2*256, true, hB, lane);
    dense_layer<8, 8>([&](int kt) { return hB[kt]; }, W + OFF_H(3), b_hid + m*1536 + 3*256, true, hA, lane);
    // skip: [emb64pad | h256] (K=320) -> 256
    dense_layer<10, 8>([&](int kt) { return kt == 0 ? embF0 : (kt == 1 ? embF1 : hA[kt - 2]); },
                       W + OFF_SKIP, b_skip + m*256, true, hB, lane);
    // L6, L7
    dense_layer<8, 8>([&](int kt) { return hB[kt]; }, W + OFF_H4, b_hid + m*1536 + 4*256, true, hA, lane);
    dense_layer<8, 8>([&](int kt) { return hA[kt]; }, W + OFF_H5, b_hid + m*1536 + 5*256, true, hB, lane);
    // alpha head (256 -> 1, padded to 16 dims): point = lane (lanes 0..15), dim0 = VGPR0
    v8f aAcc = head_tile<8>([&](int kt) { return hB[kt]; }, W + OFF_ALPHA, lane);
    // feat: 256 -> 256, NO relu
    dense_layer<8, 8>([&](int kt) { return hB[kt]; }, W + OFF_FEAT, b_feat + m*256, false, hA, lane);
    // views: [feat256 | dirs32pad] (K=288) -> 128
    dense_layer<9, 4>([&](int kt) { return kt < 8 ? hA[kt] : dirF; },
                      W + OFF_VIEWS, b_views + m*128, true, vout, lane);
    // rgb head (128 -> 3, padded to 16 dims)
    v8f rAcc = head_tile<4>([&](int kt) { return vout[kt]; }, W + OFF_RGB, lane);

    // final coalesced store: lanes 0..15 hold dims 0..7 in VGPRs 0..7 (point=lane)
    if (hl == 0 && myN >= 0) {
        const float fac = (float)m;               // reference's factor = model id
        float4 o;
        o.x = (rAcc[0] + b_rgb[m*3+0]) * fac;
        o.y = (rAcc[1] + b_rgb[m*3+1]) * fac;
        o.z = (rAcc[2] + b_rgb[m*3+2]) * fac;
        o.w = (aAcc[0] + b_alpha[m]) * fac;
        *(float4*)(out + (size_t)myN * 4) = o;
    }
}

// ---------------------------------------------------------------------------
extern "C" void kernel_launch(void* const* d_in, const int* in_sizes, int n_in,
                              void* d_out, int out_size, void* d_ws, size_t ws_size,
                              hipStream_t stream) {
    const float* pts     = (const float*)d_in[0];
    const float* dirsp   = (const float*)d_in[1];
    const float* w_in    = (const float*)d_in[2];
    const float* b_in    = (const float*)d_in[3];
    const float* w_hid   = (const float*)d_in[4];
    const float* b_hid   = (const float*)d_in[5];
    const float* w_skip  = (const float*)d_in[6];
    const float* b_skip  = (const float*)d_in[7];
    const float* w_feat  = (const float*)d_in[8];
    const float* b_feat  = (const float*)d_in[9];
    const float* w_alpha = (const float*)d_in[10];
    const float* b_alpha = (const float*)d_in[11];
    const float* w_views = (const float*)d_in[12];
    const float* b_views = (const float*)d_in[13];
    const float* w_rgb   = (const float*)d_in[14];
    const float* b_rgb   = (const float*)d_in[15];
    const int*   himg    = (const int*)d_in[16];
    float* out = (float*)d_out;

    char* ws = (char*)d_ws;
    int*    counts = (int*)ws;                                   // 256 B
    int*    lists  = (int*)(ws + 256);                           // 4*N ints
    bf16_t* packed = (bf16_t*)(ws + 256 + (size_t)4 * N_PTS * 4);// 4*600064 bf16

    init_counts<<<1, 32, 0, stream>>>(counts);
    route_zero<<<N_PTS / 256, 256, 0, stream>>>(pts, himg, out, counts, lists);

    auto packLaunch = [&](const float* src, long srcStride, long dstOfs,
                          int Ksrc, int Ncols, int KT, int nCT, int remap) {
        int elems = nCT * KT * 512;
        dim3 g((elems + 255) / 256, 4);
        pack_kernel<<<g, 256, 0, stream>>>(src, packed + dstOfs, srcStride,
                                           (long)PACK_PER_MODEL, elems,
                                           Ksrc, Ncols, KT, remap);
    };
    packLaunch(w_in,              51L * 256, OFF_L0,    51, 256, 2, 16, 0);
    packLaunch(w_hid + 0 * 65536, 6L * 65536, OFF_H(0), 256, 256, 8, 16, 0);
    packLaunch(w_hid + 1 * 65536, 6L * 65536, OFF_H(1), 256, 256, 8, 16, 0);
    packLaunch(w_hid + 2 * 65536, 6L * 65536, OFF_H(2), 256, 256, 8, 16, 0);
    packLaunch(w_hid + 3 * 65536, 6L * 65536, OFF_H(3), 256, 256, 8, 16, 0);
    packLaunch(w_hid + 4 * 65536, 6L * 65536, OFF_H4,   256, 256, 8, 16, 0);
    packLaunch(w_hid + 5 * 65536, 6L * 65536, OFF_H5,   256, 256, 8, 16, 0);
    packLaunch(w_skip,            307L * 256, OFF_SKIP, 307, 256, 10, 16, 1);
    packLaunch(w_feat,            256L * 256, OFF_FEAT, 256, 256, 8, 16, 0);
    packLaunch(w_alpha,           256L * 1,   OFF_ALPHA,256, 1,   8, 1,  0);
    packLaunch(w_views,           277L * 128, OFF_VIEWS,277, 128, 9, 8,  0);
    packLaunch(w_rgb,             128L * 3,   OFF_RGB,  128, 3,   4, 1,  0);

    size_t smem = (size_t)(128 * 64 + 128 * 32) * sizeof(bf16_t);  // 24 KB
    dim3 grid(N_PTS / TILE, 3);   // models 1..3 only; model 0 contributes zero
    mlp_kernel<<<grid, 256, smem, stream>>>(pts, dirsp, packed,
                                            b_in, b_hid, b_skip, b_feat,
                                            b_alpha, b_views, b_rgb,
                                            counts, lists, out);
}
